// DiagonalMahalanobisDistance_3058016715341
// MI455X (gfx1250) — matrix operand order repaired and checked by reading, not compiled, not run on hardware
//
#include <hip/hip_runtime.h>
#include <hip/hip_bf16.h>

// CDNA5 / gfx1250: wave32, WMMA 16x16x32 bf16 split-precision GEMM with
// gfx1250 async global->LDS staging (GLOBAL_LOAD_ASYNC_TO_LDS_B128, ASYNCcnt)
// and double-buffered LDS.
//
// d2[n,m] = ||xs_n||^2 + ||ys_m||^2 - 2 * xs @ ys^T,  xs = x*exp(log_scale)
// cross ~= hi*hi + hi*lo + lo*hi (bf16x2 split, fp32 accumulate, ~2^-18 products)

typedef __attribute__((ext_vector_type(16))) __bf16       v16bf;
typedef __attribute__((ext_vector_type(8)))  float        v8f;
typedef __attribute__((ext_vector_type(4)))  unsigned int u32x4;

#define LDS_STRIDE 40  // 32 payload + 8 pad bf16 -> 80B row stride (bank spread)

// --------------------------------------------------- gfx1250 async primitives
// INST_OFFSET is added to BOTH the LDS destination and the global address,
// and our LDS/global chunk strides are equal (16B), so one base + offset works.
#define ASYNC_LD_B128(ldsOff, gAddr, IMM)                                      \
    asm volatile("global_load_async_to_lds_b128 %0, %1, off offset:" #IMM     \
                 :: "v"(ldsOff), "v"(gAddr) : "memory")

__device__ __forceinline__ void s_wait_async0() {
    asm volatile("s_wait_asynccnt 0" ::: "memory");
}

// Low 32 bits of a generic (SHARED-aperture) pointer == LDS byte address.
__device__ __forceinline__ unsigned lds_addr(const void* p) {
    return (unsigned)(unsigned long long)reinterpret_cast<unsigned long long>((void*)p);
}

// ---------------------------------------------------------------- prep kernel
__global__ void prep_kernel(const float* __restrict__ x, const float* __restrict__ y,
                            const float* __restrict__ log_scale,
                            __bf16* __restrict__ xh, __bf16* __restrict__ xl,
                            __bf16* __restrict__ yh, __bf16* __restrict__ yl,
                            float* __restrict__ xn, float* __restrict__ yn,
                            int N, int M, int D)
{
    const int wid  = threadIdx.x >> 5;   // wave32
    const int lane = threadIdx.x & 31;
    const int row  = blockIdx.x * 8 + wid;
    if (row >= N + M) return;

    const float* src;
    __bf16 *dh, *dl;
    float* nrm;
    if (row < N) {
        src = x  + (size_t)row * D;
        dh  = xh + (size_t)row * D;
        dl  = xl + (size_t)row * D;
        nrm = xn + row;
    } else {
        const int r = row - N;
        src = y  + (size_t)r * D;
        dh  = yh + (size_t)r * D;
        dl  = yl + (size_t)r * D;
        nrm = yn + r;
    }

    float acc = 0.0f;
    for (int d = lane; d < D; d += 32) {
        const float s = __expf(log_scale[d]);
        const float v = src[d] * s;
        const __bf16 h   = (__bf16)v;
        const float  res = v - (float)h;     // exact in fp32
        dh[d] = h;
        dl[d] = (__bf16)res;
        acc += v * v;                        // norm in full fp32
    }
#pragma unroll
    for (int off = 16; off > 0; off >>= 1)
        acc += __shfl_xor(acc, off, 32);
    if (lane == 0) *nrm = acc;
}

// ------------------------------------------------------------ fragment loads
// A (16x32 bf16): lane<16 -> row=lane, K in [0..7] and [16..23];
//                 lane>=16 -> row=lane-16, K in [8..15] and [24..31].
__device__ __forceinline__ v16bf load_frag_A(const __bf16* s, int tile, int lane)
{
    const int row  = tile * 16 + (lane & 15);
    const int half = lane >> 4;
    const __bf16* base = s + row * LDS_STRIDE;
    union { u32x4 u[2]; v16bf v; } t;
    t.u[0] = *(const u32x4*)(base + 8 * half);       // K = 8*half .. +7
    t.u[1] = *(const u32x4*)(base + 8 * half + 16);  // K = 16+8*half .. +7
    return t.v;
}

// B (32x16 bf16): lane = column (= ys row); K in [8*half .. +7] and [16+8*half .. +7].
__device__ __forceinline__ v16bf load_frag_B(const __bf16* s, int tile, int lane)
{
    const int row  = tile * 16 + (lane & 15);
    const int half = lane >> 4;
    const __bf16* base = s + row * LDS_STRIDE + 8 * half;
    union { u32x4 u[2]; v16bf v; } t;
    t.u[0] = *(const u32x4*)(base);
    t.u[1] = *(const u32x4*)(base + 16);
    return t.v;
}

// ---------------------------------------------------------------- gemm kernel
__global__ __launch_bounds__(128)
void gemm_kernel(const __bf16* __restrict__ xh, const __bf16* __restrict__ xl,
                 const __bf16* __restrict__ yh, const __bf16* __restrict__ yl,
                 const float* __restrict__ xn, const float* __restrict__ yn,
                 float* __restrict__ out, int N, int M, int D)
{
    // [buffer][array: xh,xl,yh,yl][row][col]  -> 2 * 4 * 128 * 40 * 2B = 80 KB
    __shared__ __bf16 lds[2][4][128][LDS_STRIDE];

    const int t     = threadIdx.x;
    const int lane  = t & 31;
    const int wid   = t >> 5;          // 4 waves
    const int waveM = (wid >> 1) * 64; // 2x2 wave grid -> 128x128 block tile
    const int waveN = (wid & 1) * 64;
    const int bm = blockIdx.x * 128;
    const int bn = blockIdx.y * 128;

    // Per-thread global row bases (thread t owns row t of each staged array).
    const __bf16* gxh = xh + (size_t)(bm + t) * D;
    const __bf16* gxl = xl + (size_t)(bm + t) * D;
    const __bf16* gyh = yh + (size_t)(bn + t) * D;
    const __bf16* gyl = yl + (size_t)(bn + t) * D;

    // Async-stage one 128x32 k-slice of all four arrays into buffer b.
    auto stage = [&](int b, int k0) {
        const unsigned l0 = lds_addr(&lds[b][0][t][0]);
        const unsigned l1 = lds_addr(&lds[b][1][t][0]);
        const unsigned l2 = lds_addr(&lds[b][2][t][0]);
        const unsigned l3 = lds_addr(&lds[b][3][t][0]);
        const unsigned long long g0 = (unsigned long long)(gxh + k0);
        const unsigned long long g1 = (unsigned long long)(gxl + k0);
        const unsigned long long g2 = (unsigned long long)(gyh + k0);
        const unsigned long long g3 = (unsigned long long)(gyl + k0);
        ASYNC_LD_B128(l0, g0, 0);  ASYNC_LD_B128(l0, g0, 16);
        ASYNC_LD_B128(l0, g0, 32); ASYNC_LD_B128(l0, g0, 48);
        ASYNC_LD_B128(l1, g1, 0);  ASYNC_LD_B128(l1, g1, 16);
        ASYNC_LD_B128(l1, g1, 32); ASYNC_LD_B128(l1, g1, 48);
        ASYNC_LD_B128(l2, g2, 0);  ASYNC_LD_B128(l2, g2, 16);
        ASYNC_LD_B128(l2, g2, 32); ASYNC_LD_B128(l2, g2, 48);
        ASYNC_LD_B128(l3, g3, 0);  ASYNC_LD_B128(l3, g3, 16);
        ASYNC_LD_B128(l3, g3, 32); ASYNC_LD_B128(l3, g3, 48);
    };

    const v8f vzero = {0.f, 0.f, 0.f, 0.f, 0.f, 0.f, 0.f, 0.f};
    v8f acc[4][4];
#pragma unroll
    for (int r = 0; r < 4; ++r)
#pragma unroll
        for (int c = 0; c < 4; ++c)
            acc[r][c] = vzero;

    const int KT = D >> 5;   // k-steps of 32
    stage(0, 0);             // prologue: fill buffer 0

    for (int kt = 0; kt < KT; ++kt) {
        const int cur = kt & 1;

        s_wait_async0();     // my async loads into lds[cur] have landed
        __syncthreads();     // everyone's loads landed; old reads of nxt done

        if (kt + 1 < KT)     // prefetch next k-slice into the other buffer
            stage(cur ^ 1, (kt + 1) << 5);

        const __bf16* sxh = &lds[cur][0][0][0];
        const __bf16* sxl = &lds[cur][1][0][0];
        const __bf16* syh = &lds[cur][2][0][0];
        const __bf16* syl = &lds[cur][3][0][0];

        v16bf ah[4], al[4], bh[4], bl[4];
#pragma unroll
        for (int r = 0; r < 4; ++r) {
            ah[r] = load_frag_A(sxh, (waveM >> 4) + r, lane);
            al[r] = load_frag_A(sxl, (waveM >> 4) + r, lane);
        }
#pragma unroll
        for (int c = 0; c < 4; ++c) {
            bh[c] = load_frag_B(syh, (waveN >> 4) + c, lane);
            bl[c] = load_frag_B(syl, (waveN >> 4) + c, lane);
        }

#pragma unroll
        for (int r = 0; r < 4; ++r)
#pragma unroll
            for (int c = 0; c < 4; ++c) {
                acc[r][c] = __builtin_amdgcn_wmma_f32_16x16x32_bf16(
                    false, ah[r], false, bh[c], (short)0, acc[r][c], false, false);
                acc[r][c] = __builtin_amdgcn_wmma_f32_16x16x32_bf16(
                    false, ah[r], false, bl[c], (short)0, acc[r][c], false, false);
                acc[r][c] = __builtin_amdgcn_wmma_f32_16x16x32_bf16(
                    false, al[r], false, bh[c], (short)0, acc[r][c], false, false);
            }
    }

    // Epilogue: d2 = xn[m] + yn[n] - 2*cross. C/D layout: VGPR j -> row j+8*half,
    // lane&15 -> col.
    const int half = lane >> 4;
    const int nlo  = lane & 15;
#pragma unroll
    for (int r = 0; r < 4; ++r) {
        const int rowBase = bm + waveM + 16 * r + 8 * half;  // multiple of 8
        const v8f xnf = *(const v8f*)(xn + rowBase);         // 32B aligned
#pragma unroll
        for (int c = 0; c < 4; ++c) {
            const int col = bn + waveN + 16 * c + nlo;
            const float ynv = yn[col];
#pragma unroll
            for (int j = 0; j < 8; ++j) {
                out[(size_t)(rowBase + j) * (size_t)M + col] =
                    xnf[j] + ynv - 2.0f * acc[r][c][j];
            }
        }
    }
}

// ------------------------------------------------------------------ launcher
extern "C" void kernel_launch(void* const* d_in, const int* in_sizes, int n_in,
                              void* d_out, int out_size, void* d_ws, size_t ws_size,
                              hipStream_t stream)
{
    const float* x  = (const float*)d_in[0];
    const float* y  = (const float*)d_in[1];
    const float* ls = (const float*)d_in[2];
    const int D = in_sizes[2];
    const int N = in_sizes[0] / D;
    const int M = in_sizes[1] / D;

    // Workspace layout: xs_hi | xs_lo | ys_hi | ys_lo | xn | yn  (~32.06 MB)
    char* ws = (char*)d_ws;
    const size_t szX = (size_t)N * D * sizeof(__bf16);
    const size_t szY = (size_t)M * D * sizeof(__bf16);
    __bf16* xh = (__bf16*)(ws);
    __bf16* xl = (__bf16*)(ws + szX);
    __bf16* yh = (__bf16*)(ws + 2 * szX);
    __bf16* yl = (__bf16*)(ws + 2 * szX + szY);
    float*  xn = (float*)(ws + 2 * szX + 2 * szY);
    float*  yn = xn + N;

    const int rows = N + M;
    prep_kernel<<<dim3((rows + 7) / 8), dim3(256), 0, stream>>>(
        x, y, ls, xh, xl, yh, yl, xn, yn, N, M, D);

    dim3 grid(N / 128, M / 128);
    gemm_kernel<<<grid, dim3(128), 0, stream>>>(
        xh, xl, yh, yl, xn, yn, (float*)d_out, N, M, D);
}
